// MsaHmmLayer_46205258170899
// MI455X (gfx1250) — matrix operand order repaired and checked by reading, not compiled, not run on hardware
//
#include <hip/hip_runtime.h>

// HMM forward-backward, M=2, Bb=32, L=512, Q=512, S=26.
// 4 workgroups (model x batch-half16) x 512 threads (16 wave32s).
// Each wave owns TWO 16x16 output tiles: one LDS alpha-tile read feeds two
// v_wmma_f32_16x16x32_bf16 per kt (32 WMMA / wave / step, K=512), B-tiles
// dist-2 software-pipelined from L2 (2nd stream = +16KB immediate offset).
// Row sums: 16 reduction waves, stride-32 LDS reads + raw ds_bpermute tree.
// Forward stores linear scaled-alpha; backward fuses log(alpha*beta)+offset.

typedef __attribute__((ext_vector_type(16))) __bf16 v16bf;
typedef __attribute__((ext_vector_type(8)))  float  v8f;

#define Mm 2
#define BB 32
#define LL 512
#define QQ 512
#define SS 26
#define EPSV 1e-30f
#define FLOORV 1e-18f

__device__ __forceinline__ unsigned short f2bf(float f) {
  union { float f; unsigned u; } v; v.f = f;
  unsigned r = v.u + 0x7FFFu + ((v.u >> 16) & 1u);   // round-to-nearest-even
  return (unsigned short)(r >> 16);
}
__device__ __forceinline__ unsigned packbf(float a, float b) {
  return (unsigned)f2bf(a) | ((unsigned)f2bf(b) << 16);
}
// 16-bit A-operand (16x32) contiguous k order used here (k = 16*(l/16) + 2j, same for B): lane l, dword j -> K base (even)
__device__ __forceinline__ int aKlocal(int l, int j) {
  return 2 * j + ((l & 16) ? 16 : 0);
}
// load a v16bf operand from an LDS tile stored [dword j][lane l]
__device__ __forceinline__ v16bf ldsTile(const unsigned* lanePtr) {
  union { unsigned u[8]; v16bf v; } r;
#pragma unroll
  for (int j = 0; j < 8; ++j) r.u[j] = lanePtr[j * 32];
  return r.v;
}
__device__ __forceinline__ float bperm(int byteIdx, float v) {
  return __int_as_float(__builtin_amdgcn_ds_bpermute(byteIdx, __float_as_int(v)));
}

// ---- prep: swizzle A (and A^T) fp32 -> bf16 WMMA B-operand tiles in ws ----
__global__ void prep_tiles(const float* __restrict__ A, unsigned* __restrict__ ws) {
  int gid = blockIdx.x * 256 + threadIdx.x;           // 524288 total
  if (gid >= 524288) return;
  int arr = gid >> 18;                                // 0: Atile, 1: ATtile
  int id2 = gid & 262143;
  int mm  = id2 >> 17;
  int r   = id2 & 131071;
  int pt  = r >> 12;
  int kt  = (r >> 8) & 15;
  int l   = (r >> 3) & 31;
  int j   = r & 7;
  int n   = pt * 16 + (l & 15);
  int k   = kt * 32 + 2 * j + ((l & 16) ? 16 : 0);
  float v0, v1;
  if (arr == 0) {                                     // forward: B-op[k][n] = A[k][n]
    v0 = A[(mm * QQ + k) * QQ + n];
    v1 = A[(mm * QQ + k + 1) * QQ + n];
  } else {                                            // backward: B-op[k][n] = A[n][k]
    v0 = A[(mm * QQ + n) * QQ + k];
    v1 = A[(mm * QQ + n) * QQ + k + 1];
  }
  ws[gid] = packbf(v0, v1);
}

// ---- fused forward + backward + posterior combine ----
__global__ __launch_bounds__(512, 1) void fb_kernel(
    const float* __restrict__ inp, const float* __restrict__ piG,
    const float* __restrict__ Bg, const unsigned* __restrict__ wsT,
    float* __restrict__ out) {
  extern __shared__ unsigned ldsu[];
  float* ldsf = (float*)ldsu;
  // LDS offsets in dwords; operand tiles stored [dword j][lane l]
  const int BMT = 0;        // emission B^T tiles: 32 x 256 = 8192
  const int XBF = 8192;     // alpha/u bf16 A-op tiles: 16 x 256 = 4096
  const int ATMP = 12288;   // fp32 staging 16x512: 8192
  const int BETA = 20480;   // fp32 beta 16x512: 8192
  const int INPA = 28672;   // input-row A-op tile: 256
  const int CINV = 28944, LOGZ = 28960, LLIK = 28976, COMB = 28992;
  const int LOGZF = 29056;  // forward log-scale per (t,row): 512*16 = 8192
  // total 37248 dwords = 148992 B

  const int tid  = threadIdx.x;
  const int wg   = blockIdx.x;
  const int m    = wg >> 1;
  const int b0   = (wg & 1) * 16;
  const int w    = tid >> 5;     // wave id; owns p-tiles 2w and 2w+1
  const int l    = tid & 31;     // lane
  const int col  = l & 15;
  const int row0 = (l & 16) ? 8 : 0;
  const int pA   = 2 * w * 16 + col;        // column of tile A
  // base of this wave's B-tile stream; tile (2w+1) is at +4096 dwords (16KB)
  const unsigned* At  = wsT + ((size_t)(m * 32 + 2 * w) * 16) * 256;
  const unsigned* ATt = wsT + 262144 + ((size_t)(m * 32 + 2 * w) * 16) * 256;
  const size_t MBLQ = (size_t)Mm * BB * LL * QQ;

  // build emission B^T tiles: dword idx d = pt*256 + j*32 + l
  for (int i = 0; i < 16; ++i) {
    int d = tid * 16 + i;
    int pt = d >> 8, j = (d >> 5) & 7, la = d & 31;
    int k = 2 * j + ((la & 16) ? 16 : 0);
    int p = pt * 16 + (la & 15);
    float v0 = (k     < SS) ? Bg[(m * QQ + p) * SS + k]     : 0.f;
    float v1 = (k + 1 < SS) ? Bg[(m * QQ + p) * SS + k + 1] : 0.f;
    ldsu[BMT + d] = packbf(v0, v1);
  }
  if (tid < 16) ldsf[LOGZ + tid] = 0.f;
  __syncthreads();

  // ================= forward: store linear scaled-alpha =================
  for (int t = 0; t < LL; ++t) {
    if (tid < 256) {   // stage input rows (16 x 32bf16 A-op tile) for time t
      int j = (tid >> 5) & 7, la = tid & 31, r = la & 15;
      int k = aKlocal(la, j);
      size_t base = ((size_t)((m * BB + b0 + r) * LL + t)) * SS;
      float v0 = (k     < SS) ? inp[base + k]     : 0.f;
      float v1 = (k + 1 < SS) ? inp[base + k + 1] : 0.f;
      ldsu[INPA + tid] = packbf(v0, v1);
    }
    __syncthreads();

    v8f a8a, a8b;
    if (t == 0) {
      v16bf ai  = ldsTile(&ldsu[INPA + l]);
      v16bf bmA = ldsTile(&ldsu[BMT + 2 * w * 256 + l]);
      v16bf bmB = ldsTile(&ldsu[BMT + (2 * w + 1) * 256 + l]);
      v8f z = {};
      v8f e8a = __builtin_amdgcn_wmma_f32_16x16x32_bf16(false, ai, false, bmA,
                                                        (short)0, z, false, false);
      v8f e8b = __builtin_amdgcn_wmma_f32_16x16x32_bf16(false, ai, false, bmB,
                                                        (short)0, z, false, false);
      float pv0 = piG[m * QQ + pA];
      float pv1 = piG[m * QQ + pA + 16];
      for (int j = 0; j < 8; ++j) { a8a[j] = e8a[j] * pv0; a8b[j] = e8b[j] * pv1; }
    } else {
      v8f acc0 = {}, acc1 = {};
      v16bf q0a = *(const v16bf*)&At[0 * 256 + l * 8];
      v16bf q0b = *(const v16bf*)&At[4096 + 0 * 256 + l * 8];
      v16bf q1a = *(const v16bf*)&At[1 * 256 + l * 8];
      v16bf q1b = *(const v16bf*)&At[4096 + 1 * 256 + l * 8];
#pragma unroll
      for (int kt = 0; kt < 16; ++kt) {   // one xa feeds two WMMAs
        v16bf xa = ldsTile(&ldsu[XBF + kt * 256 + l]);
        acc0 = __builtin_amdgcn_wmma_f32_16x16x32_bf16(false, xa, false, q0a,
                                                       (short)0, acc0, false, false);
        acc1 = __builtin_amdgcn_wmma_f32_16x16x32_bf16(false, xa, false, q0b,
                                                       (short)0, acc1, false, false);
        q0a = q1a; q0b = q1b;
        if (kt + 2 < 16) {
          q1a = *(const v16bf*)&At[(kt + 2) * 256 + l * 8];
          q1b = *(const v16bf*)&At[4096 + (kt + 2) * 256 + l * 8];
        }
      }
      // emission WMMAs after the deep loop (short e live range)
      v16bf ai  = ldsTile(&ldsu[INPA + l]);
      v16bf bmA = ldsTile(&ldsu[BMT + 2 * w * 256 + l]);
      v16bf bmB = ldsTile(&ldsu[BMT + (2 * w + 1) * 256 + l]);
      v8f z = {};
      v8f e8a = __builtin_amdgcn_wmma_f32_16x16x32_bf16(false, ai, false, bmA,
                                                        (short)0, z, false, false);
      v8f e8b = __builtin_amdgcn_wmma_f32_16x16x32_bf16(false, ai, false, bmB,
                                                        (short)0, z, false, false);
      for (int j = 0; j < 8; ++j) { a8a[j] = acc0[j] * e8a[j]; a8b[j] = acc1[j] * e8b[j]; }
    }
    for (int j = 0; j < 8; ++j) {
      ldsf[ATMP + (row0 + j) * QQ + pA]      = a8a[j];
      ldsf[ATMP + (row0 + j) * QQ + pA + 16] = a8b[j];
    }
    __syncthreads();

    {   // all 16 waves reduce: wave w sums row w (stride-32, conflict-free)
      float s = 0.f;
      const float* rowp = &ldsf[ATMP + w * QQ + l];
#pragma unroll
      for (int i = 0; i < 16; ++i) s += rowp[i * 32];
      int lb = l * 4;
      s += bperm(lb ^ 4, s);  s += bperm(lb ^ 8, s);  s += bperm(lb ^ 16, s);
      s += bperm(lb ^ 32, s); s += bperm(lb ^ 64, s);
      if (l == 0) {
        float c = fmaxf(s, EPSV);
        ldsf[CINV + w] = 1.f / c;
        float lz = ldsf[LOGZ + w] + __logf(c);
        ldsf[LOGZ + w] = lz;
        ldsf[LOGZF + t * 16 + w] = lz;   // save forward scale for backward
      }
    }
    __syncthreads();
    // normalize, repack alpha->bf16 A-op layout, store LINEAR alpha (no logs)
    for (int i = 0; i < 8; ++i) {
      int d = tid * 8 + i;
      int kt = d >> 8, j = (d >> 5) & 7, la = d & 31;
      int r = la & 15;
      int k = kt * 32 + aKlocal(la, j);
      float ci = ldsf[CINV + r];
      float a0 = ldsf[ATMP + r * QQ + k] * ci;
      float a1 = ldsf[ATMP + r * QQ + k + 1] * ci;
      ldsu[XBF + d] = packbf(a0, a1);
      float2 o; o.x = a0; o.y = a1;
      *(float2*)&out[((size_t)((m * BB + b0 + r) * LL + t)) * QQ + k] = o;
    }
    __syncthreads();
  }

  // loglik; init beta = 1; posterior at t=L-1 = log(alpha) (logzF(L-1)==loglik)
  if (tid < 16) {
    float lk = ldsf[LOGZ + tid];
    ldsf[LLIK + tid] = lk;
    out[MBLQ + m * BB + b0 + tid] = lk;
    ldsf[LOGZ + tid] = 0.f;
  }
  __syncthreads();
  for (int i = 0; i < 16; ++i) {
    int idx = tid * 16 + i;
    int r = idx >> 9, q = idx & 511;
    ldsf[BETA + idx] = 1.f;
    size_t o = ((size_t)((m * BB + b0 + r) * LL + (LL - 1))) * QQ + q;
    out[o] = __logf(fmaxf(out[o], EPSV));
  }
  __syncthreads();

  // ================= backward + fused posterior combine =================
  for (int t = LL - 2; t >= 0; --t) {
    if (tid < 256) {   // input rows at time t+1
      int j = (tid >> 5) & 7, la = tid & 31, r = la & 15;
      int k = aKlocal(la, j);
      size_t base = ((size_t)((m * BB + b0 + r) * LL + (t + 1))) * SS;
      float v0 = (k     < SS) ? inp[base + k]     : 0.f;
      float v1 = (k + 1 < SS) ? inp[base + k + 1] : 0.f;
      ldsu[INPA + tid] = packbf(v0, v1);
    }
    __syncthreads();
    {
      v16bf ai  = ldsTile(&ldsu[INPA + l]);
      v16bf bmA = ldsTile(&ldsu[BMT + 2 * w * 256 + l]);
      v16bf bmB = ldsTile(&ldsu[BMT + (2 * w + 1) * 256 + l]);
      v8f z = {};
      v8f e8a = __builtin_amdgcn_wmma_f32_16x16x32_bf16(false, ai, false, bmA,
                                                        (short)0, z, false, false);
      v8f e8b = __builtin_amdgcn_wmma_f32_16x16x32_bf16(false, ai, false, bmB,
                                                        (short)0, z, false, false);
      for (int j = 0; j < 8; ++j) {   // u = e(t+1) * beta(t+1)
        int r = row0 + j;
        ldsf[ATMP + r * QQ + pA]      = e8a[j] * ldsf[BETA + r * QQ + pA];
        ldsf[ATMP + r * QQ + pA + 16] = e8b[j] * ldsf[BETA + r * QQ + pA + 16];
      }
    }
    __syncthreads();
    for (int i = 0; i < 8; ++i) {   // repack u -> bf16 A-op layout
      int d = tid * 8 + i;
      int kt = d >> 8, j = (d >> 5) & 7, la = d & 31;
      int r = la & 15;
      int k = kt * 32 + aKlocal(la, j);
      ldsu[XBF + d] = packbf(ldsf[ATMP + r * QQ + k], ldsf[ATMP + r * QQ + k + 1]);
    }
    __syncthreads();
    v8f acc0 = {}, acc1 = {};
    {
      v16bf q0a = *(const v16bf*)&ATt[0 * 256 + l * 8];
      v16bf q0b = *(const v16bf*)&ATt[4096 + 0 * 256 + l * 8];
      v16bf q1a = *(const v16bf*)&ATt[1 * 256 + l * 8];
      v16bf q1b = *(const v16bf*)&ATt[4096 + 1 * 256 + l * 8];
#pragma unroll
      for (int kt = 0; kt < 16; ++kt) {  // b_new = u @ A^T
        v16bf xa = ldsTile(&ldsu[XBF + kt * 256 + l]);
        acc0 = __builtin_amdgcn_wmma_f32_16x16x32_bf16(false, xa, false, q0a,
                                                       (short)0, acc0, false, false);
        acc1 = __builtin_amdgcn_wmma_f32_16x16x32_bf16(false, xa, false, q0b,
                                                       (short)0, acc1, false, false);
        q0a = q1a; q0b = q1b;
        if (kt + 2 < 16) {
          q1a = *(const v16bf*)&ATt[(kt + 2) * 256 + l * 8];
          q1b = *(const v16bf*)&ATt[4096 + (kt + 2) * 256 + l * 8];
        }
      }
    }
    for (int j = 0; j < 8; ++j) {
      ldsf[ATMP + (row0 + j) * QQ + pA]      = acc0[j];
      ldsf[ATMP + (row0 + j) * QQ + pA + 16] = acc1[j];
    }
    __syncthreads();

    {   // reduction waves
      float s = 0.f;
      const float* rowp = &ldsf[ATMP + w * QQ + l];
#pragma unroll
      for (int i = 0; i < 16; ++i) s += rowp[i * 32];
      int lb = l * 4;
      s += bperm(lb ^ 4, s);  s += bperm(lb ^ 8, s);  s += bperm(lb ^ 16, s);
      s += bperm(lb ^ 32, s); s += bperm(lb ^ 64, s);
      if (l == 0) {
        float c = fmaxf(s, EPSV);
        ldsf[CINV + w] = 1.f / c;
        float lzb = ldsf[LOGZ + w] + __logf(c);
        ldsf[LOGZ + w] = lzb;
        // scalar bracket: logz_f(t) + logz_b - loglik
        ldsf[COMB + w] = ldsf[LOGZF + t * 16 + w] + lzb - ldsf[LLIK + w];
      }
    }
    __syncthreads();
    for (int i = 0; i < 16; ++i) {   // normalize beta, one fused log/element
      int idx = tid * 16 + i;
      int r = idx >> 9, q = idx & 511;
      float b = ldsf[ATMP + idx] * ldsf[CINV + r];
      ldsf[BETA + idx] = b;
      size_t o = ((size_t)((m * BB + b0 + r) * LL + t)) * QQ + q;
      float fa = out[o];
      out[o] = __logf(fmaxf(fa, FLOORV) * fmaxf(b, FLOORV)) + ldsf[COMB + r];
    }
    __syncthreads();
  }
}

extern "C" void kernel_launch(void* const* d_in, const int* in_sizes, int n_in,
                              void* d_out, int out_size, void* d_ws, size_t ws_size,
                              hipStream_t stream) {
  const float* inp = (const float*)d_in[0];
  const float* Ag  = (const float*)d_in[1];
  const float* piG = (const float*)d_in[2];
  const float* Bg  = (const float*)d_in[3];
  unsigned* wsT = (unsigned*)d_ws;   // needs 2 MiB for A / A^T bf16 tiles
  prep_tiles<<<2048, 256, 0, stream>>>(Ag, wsT);
  fb_kernel<<<4, 512, 37248 * 4, stream>>>(inp, piG, Bg, wsT, (float*)d_out);
}
